// ResBottleneckBlock_51719996179032
// MI455X (gfx1250) — matrix operand order repaired and checked by reading, not compiled, not run on hardware
//
#include <hip/hip_runtime.h>
#include <hip/hip_bf16.h>
#include <math.h>

// ---------------------------------------------------------------------------
// Types
// ---------------------------------------------------------------------------
typedef __attribute__((ext_vector_type(16))) __bf16    v16bf;
typedef __attribute__((ext_vector_type(8)))  float     v8f;
typedef __attribute__((ext_vector_type(4)))  unsigned  u32x4;
typedef __attribute__((ext_vector_type(4)))  float     f32x4;
typedef __attribute__((ext_vector_type(8)))  int       i32x8;
typedef __attribute__((ext_vector_type(4)))  int       i32x4;

#define BB   32
#define CC   1024
#define HH   28
#define HW   784
#define GG   16
#define CG   64
#define SQ   256
#define KB_  576      // CG*9, grouped-conv K
#define BNEPS 1e-5f

union FragBF { u32x4 q[2]; v16bf v; };
union PackU  { unsigned short h[8]; u32x4 q; };

// ---------------------------------------------------------------------------
// CDNA5 async / TDM feature detection (safe fallbacks keep the build green)
// ---------------------------------------------------------------------------
#if __has_builtin(__builtin_amdgcn_global_load_async_to_lds_b128)
#define HAS_ASYNC 1
#endif
#if __has_builtin(__builtin_amdgcn_tensor_load_to_lds)
#define HAS_TDM 1
#if __has_include(<hip/amd_detail/amd_gfx1250_TDM.h>)
#define TDM_ARGS 6
#else
#define TDM_ARGS 5
#endif
#endif

__device__ __forceinline__ void async_copy16(void* lds_dst, const void* gsrc) {
#ifdef HAS_ASYNC
    __builtin_amdgcn_global_load_async_to_lds_b128(
        (__attribute__((address_space(1))) i32x4*)(unsigned long long)gsrc,
        (__attribute__((address_space(3))) i32x4*)lds_dst,
        0, 0);
#else
    *(u32x4*)lds_dst = *(const u32x4*)gsrc;
#endif
}

__device__ __forceinline__ void async_wait_all() {
#ifdef HAS_ASYNC
#if __has_builtin(__builtin_amdgcn_s_wait_asynccnt)
    __builtin_amdgcn_s_wait_asynccnt(0);
#else
    asm volatile("s_wait_asynccnt 0x0" ::: "memory");
#endif
#endif
}

__device__ __forceinline__ unsigned lds_offset(void* p) {
    return (unsigned)(unsigned long long)(__attribute__((address_space(3))) char*)p;
}

__device__ __forceinline__ unsigned short f2bf(float f) {
    union { float f; unsigned u; } v; v.f = f;
    unsigned r = v.u + 0x7FFFu + ((v.u >> 16) & 1u);
    return (unsigned short)(r >> 16);
}
__device__ __forceinline__ float bf2f(unsigned short h) {
    union { unsigned u; float f; } v; v.u = ((unsigned)h) << 16;
    return v.f;
}

// ---------------------------------------------------------------------------
// Small prep kernels
// ---------------------------------------------------------------------------
__global__ void k_zero(float* sums, unsigned* stats) {
    int i = blockIdx.x * 256 + threadIdx.x;
    if (i < BB * CC) sums[i] = 0.f;
    if (i < 8)       stats[i] = 0u;
}

__global__ void k_bnfold(const float* ga, const float* ba, const float* ma, const float* va,
                         const float* gb, const float* bbv, const float* mb, const float* vb,
                         const float* gc, const float* bc, const float* mc, const float* vc,
                         float* scl, float* bia) {
    int i = blockIdx.x * 256 + threadIdx.x;
    if (i >= 3 * CC) return;
    int set = i / CC, c = i % CC;
    const float *g, *b, *m, *v;
    if (set == 0)      { g = ga; b = ba;  m = ma; v = va; }
    else if (set == 1) { g = gb; b = bbv; m = mb; v = vb; }
    else               { g = gc; b = bc;  m = mc; v = vc; }
    float s = g[c] * rsqrtf(v[c] + BNEPS);
    scl[i] = s;
    bia[i] = b[c] - m[c] * s;
}

__global__ void k_cvt_w(const float* wa, const float* wc,
                        unsigned short* wabf, unsigned short* wcbf) {
    int i = blockIdx.x * 256 + threadIdx.x;
    if (i < CC * CC) { wabf[i] = f2bf(wa[i]); wcbf[i] = f2bf(wc[i]); }
}

// wb [co][ci][kh][kw] -> wbbf [g][n][k] with k = (kh*3+kw)*64 + ci
__global__ void k_cvt_wb(const float* wb, unsigned short* wbbf) {
    int i = blockIdx.x * 256 + threadIdx.x;
    if (i >= CC * CG * 9) return;
    int khw = i % 9;
    int ci  = (i / 9) % CG;
    int co  = i / (9 * CG);
    wbbf[(size_t)co * KB_ + khw * CG + ci] = f2bf(wb[i]);
}

// x NCHW fp32 -> xbf [b][p][c] bf16 (tiled transpose through LDS)
__global__ __launch_bounds__(256) void k_xpose(const float* __restrict__ x,
                                               unsigned short* __restrict__ xbf) {
    __shared__ unsigned short t[112 * 136];
    int b  = blockIdx.z;
    int p0 = blockIdx.y * 112;
    int c0 = blockIdx.x * 128;
    int tid = threadIdx.x;
    for (int idx = tid; idx < 128 * 28; idx += 256) {
        int c = idx / 28, seg = idx % 28;
        f32x4 v = *(const f32x4*)&x[((size_t)(b * CC + c0 + c)) * HW + p0 + seg * 4];
        int pr = seg * 4;
        t[(pr + 0) * 136 + c] = f2bf(v.x);
        t[(pr + 1) * 136 + c] = f2bf(v.y);
        t[(pr + 2) * 136 + c] = f2bf(v.z);
        t[(pr + 3) * 136 + c] = f2bf(v.w);
    }
    __syncthreads();
    for (int idx = tid; idx < 112 * 16; idx += 256) {
        int p = idx / 16, seg = idx % 16;
        u32x4 q = *(const u32x4*)&t[p * 136 + seg * 8];
        *(u32x4*)&xbf[((size_t)(b * HW + p0 + p)) * CC + c0 + seg * 8] = q;
    }
}

// ---------------------------------------------------------------------------
// Masker: 7x7 hard mask + sparsity counts
// ---------------------------------------------------------------------------
__global__ void k_mask(const float* __restrict__ x, const float* __restrict__ mkw,
                       const float* __restrict__ mkb, float* mask7, unsigned* stats) {
    __shared__ float red[256];
    int b = blockIdx.x / 49, p = blockIdx.x % 49;
    int y0 = (p / 7) * 4, x0 = (p % 7) * 4;
    int t = threadIdx.x;
    float acc = 0.f;
    for (int c = t; c < CC; c += 256) {
        float wd = mkw[c] - mkw[CC + c];
        const float* xp = x + ((size_t)(b * CC + c)) * HW + y0 * HH + x0;
        float s = 0.f;
        for (int dy = 0; dy < 4; ++dy) {
            f32x4 v = *(const f32x4*)&xp[dy * HH];
            s += v.x + v.y + v.z + v.w;
        }
        acc += wd * s;
    }
    red[t] = acc; __syncthreads();
    for (int st = 128; st > 0; st >>= 1) { if (t < st) red[t] += red[t + st]; __syncthreads(); }
    if (t == 0) {
        float diff = red[0] * (1.f / 16.f) + (mkb[0] - mkb[1]);
        float m = (diff >= 0.f) ? 1.f : 0.f;
        mask7[b * 49 + p] = m;
        if (m > 0.5f) atomicAdd(&stats[0], 1u);
    }
}

__global__ void k_dil(const float* __restrict__ mask7, unsigned* stats) {
    __shared__ int red[256];
    int b = blockIdx.x, t = threadIdx.x;
    int cnt = 0;
    for (int p = t; p < HW; p += 256) {
        int y = p / HH, xq = p % HH;
        bool on = false;
        for (int dy = -1; dy <= 1; ++dy)
            for (int dx = -1; dx <= 1; ++dx) {
                int yy = y + dy, xx = xq + dx;
                if (yy >= 0 && yy < HH && xx >= 0 && xx < HH)
                    on = on || (mask7[b * 49 + (yy >> 2) * 7 + (xx >> 2)] > 0.5f);
            }
        if (on) ++cnt;
    }
    red[t] = cnt; __syncthreads();
    for (int st = 128; st > 0; st >>= 1) { if (t < st) red[t] += red[t + st]; __syncthreads(); }
    if (t == 0) atomicAdd(&stats[1], (unsigned)red[0]);
}

// ---------------------------------------------------------------------------
// Conv A: 1x1 GEMM (M=b*784, N=1024, K=1024), fused BN+ReLU -> h1 (bf16 [b][p][c])
// Double-buffered K loop, async-to-LDS staging, one barrier per iteration.
// ---------------------------------------------------------------------------
__global__ __launch_bounds__(256)
void k_gemm_a(const unsigned short* __restrict__ xbf, const unsigned short* __restrict__ wbf,
              const float* __restrict__ scl, const float* __restrict__ bia,
              unsigned short* __restrict__ h1bf) {
    __shared__ unsigned short As[2][112 * 40];
    __shared__ unsigned short Bs[2][128 * 40];
    int b  = blockIdx.y;
    int mt = blockIdx.x >> 3, nt = blockIdx.x & 7;
    int p0 = mt * 112, n0 = nt * 128;
    int tid = threadIdx.x;
    int wid = tid >> 5, lane = tid & 31;
    int lr = lane & 15, khalf = lane >> 4;
    const size_t brow = (size_t)b * HW;

    auto stage = [&](int ks, int buf) {
        int k0 = ks * 32;
        {   // A tile: 112x32 halfs = 448 16B segments
            int row = tid >> 2, seg = tid & 3;
            async_copy16(&As[buf][row * 40 + seg * 8],
                         &xbf[(brow + p0 + row) * CC + k0 + seg * 8]);
            if (tid < 192) {
                int idx = tid + 256;
                row = idx >> 2; seg = idx & 3;
                async_copy16(&As[buf][row * 40 + seg * 8],
                             &xbf[(brow + p0 + row) * CC + k0 + seg * 8]);
            }
        }
        {   // B tile: 128x32 halfs = 512 16B segments
            int row = tid >> 2, seg = tid & 3;
            async_copy16(&Bs[buf][row * 40 + seg * 8],
                         &wbf[(size_t)(n0 + row) * CC + k0 + seg * 8]);
            int idx = tid + 256;
            row = idx >> 2; seg = idx & 3;
            async_copy16(&Bs[buf][row * 40 + seg * 8],
                         &wbf[(size_t)(n0 + row) * CC + k0 + seg * 8]);
        }
    };

    v8f zero = {0.f, 0.f, 0.f, 0.f, 0.f, 0.f, 0.f, 0.f};
    v8f acc[7];
    for (int i = 0; i < 7; ++i) acc[i] = zero;

    stage(0, 0);
    for (int ks = 0; ks < 32; ++ks) {
        int buf = ks & 1;
        async_wait_all();
        __syncthreads();
        if (ks + 1 < 32) stage(ks + 1, buf ^ 1);
        FragBF bfr;
        bfr.q[0] = *(const u32x4*)&Bs[buf][(wid * 16 + lr) * 40 + khalf * 8];
        bfr.q[1] = *(const u32x4*)&Bs[buf][(wid * 16 + lr) * 40 + khalf * 8 + 16];
#pragma unroll
        for (int msub = 0; msub < 7; ++msub) {
            FragBF afr;
            afr.q[0] = *(const u32x4*)&As[buf][(msub * 16 + lr) * 40 + khalf * 8];
            afr.q[1] = *(const u32x4*)&As[buf][(msub * 16 + lr) * 40 + khalf * 8 + 16];
            acc[msub] = __builtin_amdgcn_wmma_f32_16x16x32_bf16(
                false, afr.v, false, bfr.v, (short)0, acc[msub], false, false);
        }
    }
    int c = n0 + wid * 16 + lr;
    float s = scl[c], bi = bia[c];
#pragma unroll
    for (int msub = 0; msub < 7; ++msub) {
        int pbase = p0 + msub * 16 + khalf * 8;
#pragma unroll
        for (int r = 0; r < 8; ++r) {
            float v = acc[msub][r] * s + bi;
            v = v > 0.f ? v : 0.f;
            h1bf[(brow + pbase + r) * CC + c] = f2bf(v);
        }
    }
}

// ---------------------------------------------------------------------------
// Conv B: 3x3 grouped conv as per-(b,g) GEMM (M=784, N=64, K=576), on-the-fly
// im2col from an LDS-resident slab.  Slab staged by the Tensor Data Mover
// (one tensor_load_to_lds with LDS pad 32dw->+4dw giving pitch 72 halfs);
// weights staged with async-to-LDS copies.  Fused BN+ReLU -> h2 + SE sums.
// ---------------------------------------------------------------------------
__global__ __launch_bounds__(256)
void k_convb(const unsigned short* __restrict__ h1bf, const unsigned short* __restrict__ wbbf,
             const float* __restrict__ scl, const float* __restrict__ bia,
             unsigned short* __restrict__ h2bf, float* __restrict__ sums) {
    extern __shared__ unsigned short smem[];
    unsigned short* hs  = smem;                 // [784][72]
    unsigned short* wsb = smem + 784 * 72;      // [64][584]
    float* ssum = (float*)(smem + 784 * 72 + 64 * 584);   // [64]
    int b = blockIdx.y, g = blockIdx.x;
    int tid = threadIdx.x;
    if (tid < 64) ssum[tid] = 0.f;

#ifdef HAS_TDM
    if (tid < 32) {
        // D# group0: count=1 | lds_addr | global tile address | type=2
        unsigned long long ga =
            (unsigned long long)(const void*)&h1bf[((size_t)b * HW) * CC + g * CG];
        u32x4 g0;
        g0.x = 1u;
        g0.y = lds_offset(hs);
        g0.z = (unsigned)ga;
        g0.w = (unsigned)((ga >> 32) & 0x1FFFFFFull) | (2u << 30);
        // D# group1: data_size=2B, pad_enable, pad_interval=32dw(code4),
        // pad_amount=4dw(code3); tensor 1024x784 halfs, tile 64x784,
        // dim0 stride = 1024 halfs.
        i32x8 g1;
        g1[0] = (int)((1u << 16) | (1u << 20) | (4u << 22) | (3u << 25));
        g1[1] = (int)(1024u << 16);   // tensor_dim0 = 1024 -> bits[79:48]
        g1[2] = (int)(784u  << 16);   // tensor_dim1 = 784  -> bits[111:80]
        g1[3] = (int)(64u   << 16);   // tile_dim0  = 64    -> bits[127:112]
        g1[4] = (int)784u;            // tile_dim1  = 784, tile_dim2 = 0
        g1[5] = (int)1024u;           // tensor_dim0_stride = 1024
        g1[6] = 0;
        g1[7] = 0;
        i32x4 zz = {0, 0, 0, 0};
#if TDM_ARGS == 6
        i32x8 z8 = {0, 0, 0, 0, 0, 0, 0, 0};
        __builtin_amdgcn_tensor_load_to_lds(g0, g1, zz, zz, z8, 0);
#else
        __builtin_amdgcn_tensor_load_to_lds(g0, g1, zz, zz, 0);
#endif
    }
#else
    // fallback: copy slab (784x64 halfs) with (async) 16B segments
    for (int i = 0; i < 24; ++i) {
        int idx = tid + i * 256;
        int p = idx >> 3, seg = idx & 7;
        async_copy16(&hs[p * 72 + seg * 8],
                     &h1bf[((size_t)(b * HW + p)) * CC + g * CG + seg * 8]);
    }
    if (tid < 128) {
        int idx = tid + 6144;
        int p = idx >> 3, seg = idx & 7;
        async_copy16(&hs[p * 72 + seg * 8],
                     &h1bf[((size_t)(b * HW + p)) * CC + g * CG + seg * 8]);
    }
#endif
    // group weights 64x576 halfs = 4608 16B segments
    for (int i = 0; i < 18; ++i) {
        int idx = tid + i * 256;
        int n = idx / 72, seg = idx % 72;
        async_copy16(&wsb[n * 584 + seg * 8],
                     &wbbf[(size_t)(g * CG + n) * KB_ + seg * 8]);
    }
    async_wait_all();
#ifdef HAS_TDM
    if (tid < 32) __builtin_amdgcn_s_wait_tensorcnt(0);
#endif
    __syncthreads();

    int wid = tid >> 5, lane = tid & 31;
    int lr = lane & 15, khalf = lane >> 4;
    u32x4 zq = {0u, 0u, 0u, 0u};
    v8f zero = {0.f, 0.f, 0.f, 0.f, 0.f, 0.f, 0.f, 0.f};

    for (int tile = wid; tile < 196; tile += 8) {           // 49 Msub x 4 Nsub
        int msub = tile >> 2, nsub = tile & 3;
        v8f acc = zero;
        int p = msub * 16 + lr;
        int y = p / HH, xq = p % HH;
#pragma unroll
        for (int ks = 0; ks < 18; ++ks) {
            int khw = ks >> 1;
            int ci0 = ((ks & 1) << 5) + khalf * 8;
            int dy = khw / 3 - 1, dx = khw % 3 - 1;
            int yy = y + dy, xx = xq + dx;
            FragBF afr;
            if (yy >= 0 && yy < HH && xx >= 0 && xx < HH) {
                afr.q[0] = *(const u32x4*)&hs[(yy * HH + xx) * 72 + ci0];
                afr.q[1] = *(const u32x4*)&hs[(yy * HH + xx) * 72 + ci0 + 16];
            } else { afr.q[0] = zq; afr.q[1] = zq; }
            FragBF bfr;
            bfr.q[0] = *(const u32x4*)&wsb[(nsub * 16 + lr) * 584 + ks * 32 + khalf * 8];
            bfr.q[1] = *(const u32x4*)&wsb[(nsub * 16 + lr) * 584 + ks * 32 + khalf * 8 + 16];
            acc = __builtin_amdgcn_wmma_f32_16x16x32_bf16(
                false, afr.v, false, bfr.v, (short)0, acc, false, false);
        }
        int cl = nsub * 16 + lr;
        int c  = g * CG + cl;
        float s = scl[c], bi = bia[c];
        int pbase = msub * 16 + khalf * 8;
        float lsum = 0.f;
#pragma unroll
        for (int r = 0; r < 8; ++r) {
            float v = acc[r] * s + bi;
            v = v > 0.f ? v : 0.f;
            lsum += v;
            h2bf[((size_t)(b * HW + pbase + r)) * CC + c] = f2bf(v);
        }
        atomicAdd(&ssum[cl], lsum);
    }
    __syncthreads();
    if (tid < 64) atomicAdd(&sums[b * CC + g * CG + tid], ssum[tid]);
}

// ---------------------------------------------------------------------------
// SE: per-batch squeeze-excite gates
// ---------------------------------------------------------------------------
__global__ __launch_bounds__(256)
void k_se(const float* __restrict__ sums, const float* __restrict__ sw1, const float* __restrict__ sb1,
          const float* __restrict__ sw2, const float* __restrict__ sb2, float* __restrict__ sbuf) {
    __shared__ float mean_s[CC];
    __shared__ float s1[SQ];
    int b = blockIdx.x, t = threadIdx.x;
    for (int c = t; c < CC; c += 256) mean_s[c] = sums[b * CC + c] * (1.f / 784.f);
    __syncthreads();
    {
        float a = sb1[t];
        const float* w = &sw1[(size_t)t * CC];
        for (int c = 0; c < CC; ++c) a += w[c] * mean_s[c];
        s1[t] = a > 0.f ? a : 0.f;
    }
    __syncthreads();
    for (int c = t; c < CC; c += 256) {
        float a = sb2[c];
        const float* w = &sw2[(size_t)c * SQ];
        for (int j = 0; j < SQ; ++j) a += w[j] * s1[j];
        sbuf[b * CC + c] = 1.f / (1.f + expf(-a));
    }
}

// ---------------------------------------------------------------------------
// Conv C: 1x1 GEMM with SE gate folded into A-tile load (sync, transforms),
// async B staging; epilogue = BN, mask, residual add (fp32 x), ReLU.
// ---------------------------------------------------------------------------
__global__ __launch_bounds__(256)
void k_gemm_c(const unsigned short* __restrict__ h2bf, const unsigned short* __restrict__ wbf,
              const float* __restrict__ sbuf, const float* __restrict__ scl,
              const float* __restrict__ bia, const float* __restrict__ x,
              const float* __restrict__ mask7, float* __restrict__ out) {
    __shared__ unsigned short As[2][112 * 40];
    __shared__ unsigned short Bs[2][128 * 40];
    __shared__ float mk[49];
    int b  = blockIdx.y;
    int mt = blockIdx.x >> 3, nt = blockIdx.x & 7;
    int p0 = mt * 112, n0 = nt * 128;
    int tid = threadIdx.x;
    if (tid < 49) mk[tid] = mask7[b * 49 + tid];
    int wid = tid >> 5, lane = tid & 31;
    int lr = lane & 15, khalf = lane >> 4;
    const size_t brow = (size_t)b * HW;

    auto stage = [&](int ks, int buf) {
        int k0 = ks * 32;
        {   // B tile (plain copy -> async)
            int row = tid >> 2, seg = tid & 3;
            async_copy16(&Bs[buf][row * 40 + seg * 8],
                         &wbf[(size_t)(n0 + row) * CC + k0 + seg * 8]);
            int idx = tid + 256;
            row = idx >> 2; seg = idx & 3;
            async_copy16(&Bs[buf][row * 40 + seg * 8],
                         &wbf[(size_t)(n0 + row) * CC + k0 + seg * 8]);
        }
#pragma unroll
        for (int pass = 0; pass < 2; ++pass) {   // A tile with SE gate (sync)
            int idx = tid + pass * 256;
            if (idx < 448) {
                int row = idx >> 2, seg = idx & 3;
                int kk = k0 + seg * 8;
                u32x4 q = *(const u32x4*)&h2bf[(brow + p0 + row) * CC + kk];
                const float* sv = &sbuf[b * CC + kk];
                const unsigned short* hp = (const unsigned short*)&q;
                PackU ou;
#pragma unroll
                for (int j = 0; j < 8; ++j) ou.h[j] = f2bf(bf2f(hp[j]) * sv[j]);
                *(u32x4*)&As[buf][row * 40 + seg * 8] = ou.q;
            }
        }
    };

    v8f zero = {0.f, 0.f, 0.f, 0.f, 0.f, 0.f, 0.f, 0.f};
    v8f acc[7];
    for (int i = 0; i < 7; ++i) acc[i] = zero;

    stage(0, 0);
    for (int ks = 0; ks < 32; ++ks) {
        int buf = ks & 1;
        async_wait_all();
        __syncthreads();
        if (ks + 1 < 32) stage(ks + 1, buf ^ 1);
        FragBF bfr;
        bfr.q[0] = *(const u32x4*)&Bs[buf][(wid * 16 + lr) * 40 + khalf * 8];
        bfr.q[1] = *(const u32x4*)&Bs[buf][(wid * 16 + lr) * 40 + khalf * 8 + 16];
#pragma unroll
        for (int msub = 0; msub < 7; ++msub) {
            FragBF afr;
            afr.q[0] = *(const u32x4*)&As[buf][(msub * 16 + lr) * 40 + khalf * 8];
            afr.q[1] = *(const u32x4*)&As[buf][(msub * 16 + lr) * 40 + khalf * 8 + 16];
            acc[msub] = __builtin_amdgcn_wmma_f32_16x16x32_bf16(
                false, afr.v, false, bfr.v, (short)0, acc[msub], false, false);
        }
    }
    int c = n0 + wid * 16 + lr;
    float s = scl[c], bi = bia[c];
#pragma unroll
    for (int msub = 0; msub < 7; ++msub) {
        int pbase = p0 + msub * 16 + khalf * 8;
        const float* xr = &x[((size_t)(b * CC + c)) * HW + pbase];
        float* orow     = &out[((size_t)(b * CC + c)) * HW + pbase];
        float vv[8];
#pragma unroll
        for (int r = 0; r < 8; ++r) {
            int p = pbase + r;
            int yy = p / HH, xx = p % HH;
            float m = mk[(yy >> 2) * 7 + (xx >> 2)];
            float v = acc[msub][r] * s + bi;
            v = v * m + xr[r];
            vv[r] = v > 0.f ? v : 0.f;
        }
        *(f32x4*)&orow[0] = *(const f32x4*)&vv[0];
        *(f32x4*)&orow[4] = *(const f32x4*)&vv[4];
    }
}

// ---------------------------------------------------------------------------
// Final scalars
// ---------------------------------------------------------------------------
__global__ void k_stats(const unsigned* stats, float* out4) {
    if (threadIdx.x != 0 || blockIdx.x != 0) return;
    float sp = (float)stats[0] / (float)(BB * 49);
    float sd = (float)stats[1] / (float)(BB * HW);
    double c1 = 1048576.0, c2 = 589824.0, c3 = 1048576.0, mf = 200704.0;
    double dense  = mf + (c1 + c2 + c3) * 784.0;
    double sparse = mf + c1 * 784.0 * (double)sd + (c2 + c3) * 784.0 * (double)sp;
    out4[0] = sp;
    out4[1] = sd;
    out4[2] = (float)(sparse / dense);
    out4[3] = (float)sparse;
}

// ---------------------------------------------------------------------------
// Launch
// ---------------------------------------------------------------------------
extern "C" void kernel_launch(void* const* d_in, const int* in_sizes, int n_in,
                              void* d_out, int out_size, void* d_ws, size_t ws_size,
                              hipStream_t stream) {
    (void)in_sizes; (void)n_in; (void)out_size; (void)ws_size;
    const float* x   = (const float*)d_in[0];
    const float* wa  = (const float*)d_in[2];
    const float* ga  = (const float*)d_in[3];
    const float* ba  = (const float*)d_in[4];
    const float* ma  = (const float*)d_in[5];
    const float* va  = (const float*)d_in[6];
    const float* wb  = (const float*)d_in[7];
    const float* gb  = (const float*)d_in[8];
    const float* bbv = (const float*)d_in[9];
    const float* mb  = (const float*)d_in[10];
    const float* vb  = (const float*)d_in[11];
    const float* sw1 = (const float*)d_in[12];
    const float* sb1 = (const float*)d_in[13];
    const float* sw2 = (const float*)d_in[14];
    const float* sb2 = (const float*)d_in[15];
    const float* wc  = (const float*)d_in[16];
    const float* gc  = (const float*)d_in[17];
    const float* bc  = (const float*)d_in[18];
    const float* mc  = (const float*)d_in[19];
    const float* vc  = (const float*)d_in[20];
    const float* mkw = (const float*)d_in[21];
    const float* mkb = (const float*)d_in[22];

    char* ws = (char*)d_ws;
    size_t off = 0;
    auto alloc = [&](size_t bytes) -> char* {
        char* p = ws + off;
        off = (off + bytes + 255) & ~(size_t)255;
        return p;
    };
    const size_t actbytes = (size_t)BB * HW * CC * 2;
    unsigned short* xbf  = (unsigned short*)alloc(actbytes);
    unsigned short* h1bf = (unsigned short*)alloc(actbytes);
    unsigned short* h2bf = (unsigned short*)alloc(actbytes);
    unsigned short* wabf = (unsigned short*)alloc((size_t)CC * CC * 2);
    unsigned short* wcbf = (unsigned short*)alloc((size_t)CC * CC * 2);
    unsigned short* wbbf = (unsigned short*)alloc((size_t)CC * KB_ * 2);
    float* scl   = (float*)alloc(3 * CC * 4);
    float* bia   = (float*)alloc(3 * CC * 4);
    float* sums  = (float*)alloc((size_t)BB * CC * 4);
    float* sbuf  = (float*)alloc((size_t)BB * CC * 4);
    float* mask7 = (float*)alloc(BB * 49 * 4);
    unsigned* stats = (unsigned*)alloc(64);

    float* out = (float*)d_out;
    const size_t NOUT = (size_t)BB * CC * HW;

    k_zero  <<<128, 256, 0, stream>>>(sums, stats);
    k_bnfold<<<12, 256, 0, stream>>>(ga, ba, ma, va, gb, bbv, mb, vb, gc, bc, mc, vc, scl, bia);
    k_cvt_w <<<4096, 256, 0, stream>>>(wa, wc, wabf, wcbf);
    k_cvt_wb<<<(CC * CG * 9 + 255) / 256, 256, 0, stream>>>(wb, wbbf);
    k_xpose <<<dim3(8, 7, 32), 256, 0, stream>>>(x, xbf);
    k_mask  <<<BB * 49, 256, 0, stream>>>(x, mkw, mkb, mask7, stats);
    k_dil   <<<BB, 256, 0, stream>>>(mask7, stats);

    k_gemm_a<<<dim3(56, 32), 256, 0, stream>>>(xbf, wabf, scl + 0 * CC, bia + 0 * CC, h1bf);

    size_t convb_lds = (size_t)784 * 72 * 2 + (size_t)64 * 584 * 2 + 64 * 4;
    k_convb<<<dim3(GG, BB), 256, convb_lds, stream>>>(h1bf, wbbf, scl + 1 * CC, bia + 1 * CC,
                                                      h2bf, sums);

    k_se    <<<BB, 256, 0, stream>>>(sums, sw1, sb1, sw2, sb2, sbuf);

    k_gemm_c<<<dim3(56, 32), 256, 0, stream>>>(h2bf, wcbf, sbuf, scl + 2 * CC, bia + 2 * CC,
                                               x, mask7, out);

    k_stats<<<1, 64, 0, stream>>>(stats, out + NOUT);
}
